// NemotronHMOE_21723944583770
// MI455X (gfx1250) — matrix operand order repaired and checked by reading, not compiled, not run on hardware
//
#include <hip/hip_runtime.h>

typedef __bf16 bf16_t;
typedef __attribute__((ext_vector_type(4)))  __bf16 v4bf;
typedef __attribute__((ext_vector_type(8)))  __bf16 v8bf;
typedef __attribute__((ext_vector_type(16))) __bf16 v16bf;
typedef __attribute__((ext_vector_type(8)))  float  v8f;

#define T_TOK 2048
#define HID   1024
#define INTER 512
#define NEXP  32
#define SINTER 2048
#define HPAD  552    // h LDS row stride (elems): 1104B = 69*16, conflict-free
#define XPAD  1048   // x LDS row stride (elems): 2096B = 131*16, conflict-free

// ---------------- fragment loaders (ISA 7.12.2 layouts) ----------------
// A-matrix 16x32 bf16: lane m (0-15): K {0..7} then {16..23}; lane m+16: K {8..15},{24..31}
__device__ __forceinline__ v16bf load_a_frag(const bf16_t* base, int ld, int lane) {
  const bf16_t* p = base + (size_t)(lane & 15) * ld + ((lane >> 4) << 3);
  v8bf lo = *(const v8bf*)(p);
  v8bf hi = *(const v8bf*)(p + 16);
  return __builtin_shufflevector(lo, hi, 0,1,2,3,4,5,6,7,8,9,10,11,12,13,14,15);
}
// B-matrix 32x16 bf16: lane n (0-15): K {0..7} then {16..23}; lane n+16: K {8..15},{24..31}
__device__ __forceinline__ v16bf load_b_frag(const bf16_t* base, int ld, int lane) {
  const bf16_t* p = base + (size_t)(lane & 15) * ld + ((lane >> 4) << 3);
  v8bf lo = *(const v8bf*)(p);
  v8bf hi = *(const v8bf*)(p + 16);
  return __builtin_shufflevector(lo, hi, 0,1,2,3,4,5,6,7,8,9,10,11,12,13,14,15);
}
__device__ __forceinline__ v8f wmma_bf16(v16bf a, v16bf b, v8f c) {
  return __builtin_amdgcn_wmma_f32_16x16x32_bf16(false, a, false, b, (short)0, c, false, false);
}

// ---------------- fp32 -> bf16 convert ----------------
__global__ void cvt_f32_bf16(const float* __restrict__ src, bf16_t* __restrict__ dst, int n) {
  int i = (blockIdx.x * blockDim.x + threadIdx.x) * 4;
  if (i >= n) return;
  float4 v = *(const float4*)(src + i);
  v4bf o;
  o.x = (bf16_t)v.x; o.y = (bf16_t)v.y; o.z = (bf16_t)v.z; o.w = (bf16_t)v.w;
  *(v4bf*)(dst + i) = o;
}

// ---------------- router: one wave per token, lane = expert ----------------
__global__ __launch_bounds__(256) void router_kernel(
    const float* __restrict__ x, const float* __restrict__ gate_w,
    const float* __restrict__ e_bias, float* __restrict__ combine) {
  int wave = threadIdx.x >> 5;
  int lane = threadIdx.x & 31;
  int t = blockIdx.x * 8 + wave;
  const float* xr = x + (size_t)t * HID;
  const float* gr = gate_w + (size_t)lane * HID;
  float logit = 0.f;
  for (int h = 0; h < HID; h += 4) {
    float4 xv = *(const float4*)(xr + h);
    float4 gv = *(const float4*)(gr + h);
    logit += xv.x*gv.x + xv.y*gv.y + xv.z*gv.z + xv.w*gv.w;
  }
  float score = 1.f / (1.f + __expf(-logit));
  float sc = score + e_bias[lane];
  // group top-2 sum within quads of lanes (8 groups x 4 experts)
  int qb = lane & ~3;
  float v0 = __shfl(sc, qb + 0, 32), v1 = __shfl(sc, qb + 1, 32);
  float v2 = __shfl(sc, qb + 2, 32), v3 = __shfl(sc, qb + 3, 32);
  float a_ = fmaxf(v0, v1), b_ = fminf(v0, v1);
  float c_ = fmaxf(v2, v3), d_ = fminf(v2, v3);
  float gsum = fmaxf(a_, c_) + fmaxf(fminf(a_, c_), fmaxf(b_, d_));
  // rank my group among 8 (ties -> lower index wins, as jax top_k)
  int myg = lane >> 2, grank = 0;
  #pragma unroll
  for (int g2 = 0; g2 < 8; ++g2) {
    float og = __shfl(gsum, g2 * 4, 32);
    grank += (og > gsum) || (og == gsum && g2 < myg);
  }
  float masked = (grank < 4) ? sc : 0.0f;
  // top-6 of 32 experts by rank
  int erank = 0;
  #pragma unroll
  for (int j = 0; j < 32; ++j) {
    float ov = __shfl(masked, j, 32);
    erank += (ov > masked) || (ov == masked && j < lane);
  }
  bool sel = erank < 6;
  float w = sel ? score : 0.0f;
  float wsum = w;
  #pragma unroll
  for (int off = 16; off; off >>= 1) wsum += __shfl_xor(wsum, off, 32);
  combine[(size_t)t * NEXP + lane] = sel ? (score / (wsum + 1e-20f)) * 2.5f : 0.0f;
}

// ---------------- MoE experts: fused up->relu2*w->down, 8 experts/group ----------------
// M=64 tokens per WG, 16 waves. mr = wave&3 (row tile), wq = wave>>2 (col group).
// X tile staged once into LDS (128 KB), reused across all 8 experts as the up GEMM's A operand.
__global__ __launch_bounds__(512) void moe_kernel(
    const bf16_t* __restrict__ xbf, const bf16_t* __restrict__ upw,
    const bf16_t* __restrict__ dnw, const float* __restrict__ combine,
    float* __restrict__ partials) {
  const int t0 = blockIdx.x * 64;
  const int g  = blockIdx.y;
  float* part = partials + (size_t)g * T_TOK * HID;
  const int wave = threadIdx.x >> 5;
  const int lane = threadIdx.x & 31;
  const int mr   = wave & 3;            // row tile 0..3 (16 tokens each)
  const int wq   = wave >> 2;           // 0..3 column group

  __shared__ bf16_t xtile[64 * XPAD];   // 134144 B
  __shared__ bf16_t hbuf[64 * HPAD];    //  70656 B
  __shared__ float  combw[64];

  // ---- stage X tile: 64 rows x 512 cols bf16, coalesced b128 copies ----
  for (int idx = threadIdx.x; idx < 64 * 64; idx += 512) {
    int row = idx >> 6;
    int c8  = (idx & 63) << 3;          // element offset within row (16B chunks)
    *(v8bf*)(xtile + row * XPAD + c8) =
        *(const v8bf*)(xbf + (size_t)(t0 + row) * HID + c8);
  }

  v8f dacc[16];
  #pragma unroll
  for (int j = 0; j < 16; ++j) dacc[j] = (v8f){};

  const int hi8  = (lane >> 4) << 3;
  const int coln = lane & 15;

  for (int ei = 0; ei < 8; ++ei) {
    const int e = g * 8 + ei;
    __syncthreads();   // xtile staged (first iter); prior hbuf/combw readers done
    if (threadIdx.x < 64)
      combw[threadIdx.x] = combine[(size_t)(t0 + threadIdx.x) * NEXP + e];

    // ---- up: h = x @ up_w[e]^T ; wave covers 8 I-tiles; A from LDS ----
    v8f uacc[8];
    #pragma unroll
    for (int j = 0; j < 8; ++j) uacc[j] = (v8f){};
    const bf16_t* xrow = xtile + mr * 16 * XPAD;
    const bf16_t* wrow = upw + (size_t)e * INTER * HID;
    for (int ks = 0; ks < HID; ks += 32) {
      v16bf a = load_a_frag(xrow + ks, XPAD, lane);
      #pragma unroll
      for (int j = 0; j < 8; ++j) {
        int n0 = (wq * 8 + j) * 16;
        v16bf b = load_b_frag(wrow + (size_t)n0 * HID + ks, HID, lane);
        uacc[j] = wmma_bf16(a, b, uacc[j]);
      }
    }
    __syncthreads();   // combw visible; prior hbuf reads finished

    // relu2 * combine -> bf16 h in LDS
    #pragma unroll
    for (int j = 0; j < 8; ++j) {
      int n0 = (wq * 8 + j) * 16;
      #pragma unroll
      for (int r = 0; r < 8; ++r) {
        int row = mr * 16 + r + hi8;
        float v = fmaxf(uacc[j][r], 0.0f);
        v = v * v * combw[row];
        hbuf[row * HPAD + n0 + coln] = (bf16_t)v;
      }
    }
    __syncthreads();   // h ready

    // ---- down: acc += h @ down_w[e]^T ; wave covers 16 H-tiles ----
    const bf16_t* hrow = hbuf + mr * 16 * HPAD;
    const bf16_t* drow = dnw + (size_t)e * HID * INTER;
    for (int ks = 0; ks < INTER; ks += 32) {
      v16bf a = load_a_frag(hrow + ks, HPAD, lane);
      #pragma unroll
      for (int j = 0; j < 16; ++j) {
        int n0 = (wq * 16 + j) * 16;
        v16bf b = load_b_frag(drow + (size_t)n0 * INTER + ks, INTER, lane);
        dacc[j] = wmma_bf16(a, b, dacc[j]);
      }
    }
  }

  #pragma unroll
  for (int j = 0; j < 16; ++j) {
    int col = (wq * 16 + j) * 16 + coln;
    #pragma unroll
    for (int r = 0; r < 8; ++r) {
      int t = t0 + mr * 16 + r + hi8;
      part[(size_t)t * HID + col] = dacc[j][r];
    }
  }
}

// ---------------- shared expert up: s = relu2(x @ su^T) -> bf16 ----------------
__global__ __launch_bounds__(256) void sup_kernel(
    const bf16_t* __restrict__ xbf, const bf16_t* __restrict__ suw,
    bf16_t* __restrict__ sact) {
  const int t0 = blockIdx.x * 32;
  const int nb = blockIdx.y * 128;
  const int wave = threadIdx.x >> 5, lane = threadIdx.x & 31;
  const int mr = wave & 1, jb = (wave >> 1) * 2;
  v8f acc[2]; acc[0] = (v8f){}; acc[1] = (v8f){};
  const bf16_t* xrow = xbf + (size_t)(t0 + mr * 16) * HID;
  for (int ks = 0; ks < HID; ks += 32) {
    v16bf a = load_a_frag(xrow + ks, HID, lane);
    #pragma unroll
    for (int j = 0; j < 2; ++j) {
      int n0 = nb + (jb + j) * 16;
      v16bf b = load_b_frag(suw + (size_t)n0 * HID + ks, HID, lane);
      acc[j] = wmma_bf16(a, b, acc[j]);
    }
  }
  const int hi8 = (lane >> 4) << 3, coln = lane & 15;
  #pragma unroll
  for (int j = 0; j < 2; ++j) {
    int col = nb + (jb + j) * 16 + coln;
    #pragma unroll
    for (int r = 0; r < 8; ++r) {
      int t = t0 + mr * 16 + r + hi8;
      float v = fmaxf(acc[j][r], 0.0f);
      sact[(size_t)t * SINTER + col] = (bf16_t)(v * v);
    }
  }
}

// ---------------- shared expert down: partial[4] = s @ sd^T ----------------
__global__ __launch_bounds__(256) void sdown_kernel(
    const bf16_t* __restrict__ sact, const bf16_t* __restrict__ sdw,
    float* __restrict__ part4) {
  const int t0 = blockIdx.x * 32;
  const int nb = blockIdx.y * 128;
  const int wave = threadIdx.x >> 5, lane = threadIdx.x & 31;
  const int mr = wave & 1, jb = (wave >> 1) * 2;
  v8f acc[2]; acc[0] = (v8f){}; acc[1] = (v8f){};
  const bf16_t* arow = sact + (size_t)(t0 + mr * 16) * SINTER;
  for (int ks = 0; ks < SINTER; ks += 32) {
    v16bf a = load_a_frag(arow + ks, SINTER, lane);
    #pragma unroll
    for (int j = 0; j < 2; ++j) {
      int n0 = nb + (jb + j) * 16;
      v16bf b = load_b_frag(sdw + (size_t)n0 * SINTER + ks, SINTER, lane);
      acc[j] = wmma_bf16(a, b, acc[j]);
    }
  }
  const int hi8 = (lane >> 4) << 3, coln = lane & 15;
  #pragma unroll
  for (int j = 0; j < 2; ++j) {
    int col = nb + (jb + j) * 16 + coln;
    #pragma unroll
    for (int r = 0; r < 8; ++r) {
      int t = t0 + mr * 16 + r + hi8;
      part4[(size_t)t * HID + col] = acc[j][r];
    }
  }
}

// ---------------- final: sum 5 partials ----------------
__global__ void final_sum(const float* __restrict__ parts, float* __restrict__ out) {
  size_t i = ((size_t)blockIdx.x * 256 + threadIdx.x) * 4;
  const size_t TH = (size_t)T_TOK * HID;
  float4 s = *(const float4*)(parts + i);
  #pragma unroll
  for (int g = 1; g < 5; ++g) {
    float4 p = *(const float4*)(parts + g * TH + i);
    s.x += p.x; s.y += p.y; s.z += p.z; s.w += p.w;
  }
  *(float4*)(out + i) = s;
}

extern "C" void kernel_launch(void* const* d_in, const int* in_sizes, int n_in,
                              void* d_out, int out_size, void* d_ws, size_t ws_size,
                              hipStream_t stream) {
  const float* x   = (const float*)d_in[0];
  const float* gw  = (const float*)d_in[1];
  const float* eb  = (const float*)d_in[2];
  const float* upw = (const float*)d_in[3];
  const float* dnw = (const float*)d_in[4];
  const float* suw = (const float*)d_in[5];
  const float* sdw = (const float*)d_in[6];
  float* out = (float*)d_out;

  char* ws = (char*)d_ws;
  const size_t OFF_X    = 0;                         // 2048*1024 bf16     = 4 MiB
  const size_t OFF_UP   = OFF_X    + 4194304;        // 32*512*1024 bf16   = 32 MiB
  const size_t OFF_DN   = OFF_UP   + 33554432;       // 32*1024*512 bf16   = 32 MiB
  const size_t OFF_SU   = OFF_DN   + 33554432;       // 2048*1024 bf16     = 4 MiB
  const size_t OFF_SD   = OFF_SU   + 4194304;        // 1024*2048 bf16     = 4 MiB
  const size_t OFF_CMB  = OFF_SD   + 4194304;        // 2048*32 f32
  const size_t OFF_SACT = OFF_CMB  + 262144;         // 2048*2048 bf16     = 8 MiB
  const size_t OFF_PART = OFF_SACT + 8388608;        // 5*2048*1024 f32    = 40 MiB

  bf16_t* xbf  = (bf16_t*)(ws + OFF_X);
  bf16_t* upbf = (bf16_t*)(ws + OFF_UP);
  bf16_t* dnbf = (bf16_t*)(ws + OFF_DN);
  bf16_t* subf = (bf16_t*)(ws + OFF_SU);
  bf16_t* sdbf = (bf16_t*)(ws + OFF_SD);
  float*  cmb  = (float*)(ws + OFF_CMB);
  bf16_t* sact = (bf16_t*)(ws + OFF_SACT);
  float*  part = (float*)(ws + OFF_PART);

  // fp32 -> bf16 conversions (per-launch, deterministic)
  cvt_f32_bf16<<<2048,  256, 0, stream>>>(x,   xbf,  T_TOK * HID);
  cvt_f32_bf16<<<16384, 256, 0, stream>>>(upw, upbf, NEXP * INTER * HID);
  cvt_f32_bf16<<<16384, 256, 0, stream>>>(dnw, dnbf, NEXP * HID * INTER);
  cvt_f32_bf16<<<2048,  256, 0, stream>>>(suw, subf, SINTER * HID);
  cvt_f32_bf16<<<2048,  256, 0, stream>>>(sdw, sdbf, HID * SINTER);

  router_kernel<<<T_TOK / 8, 256, 0, stream>>>(x, gw, eb, cmb);

  moe_kernel<<<dim3(T_TOK / 64, 4), 512, 0, stream>>>(xbf, upbf, dnbf, cmb, part);
  sup_kernel<<<dim3(T_TOK / 32, SINTER / 128), 256, 0, stream>>>(xbf, subf, sact);
  sdown_kernel<<<dim3(T_TOK / 32, HID / 128), 256, 0, stream>>>(
      sact, sdbf, part + 4ull * T_TOK * HID);

  final_sum<<<2048, 256, 0, stream>>>(part, out);
}
